// AttentionLayer_90563680403579
// MI455X (gfx1250) — matrix operand order repaired and checked by reading, not compiled, not run on hardware
//
#include <hip/hip_runtime.h>
#include <hip/hip_bf16.h>
#include <stdint.h>

// ---------- types ----------
typedef __attribute__((ext_vector_type(16))) __bf16 v16bf;
typedef __attribute__((ext_vector_type(8)))  float  v8f;
typedef uint32_t u32x4 __attribute__((ext_vector_type(4), may_alias));
typedef uint32_t u32a  __attribute__((may_alias));

#define WMMA_BF16(a, b, c) \
  __builtin_amdgcn_wmma_f32_16x16x32_bf16(false, (a), false, (b), (short)0, (c), false, false)

static constexpr int B_  = 8;
static constexpr int T_  = 1024;
static constexpr int D_  = 1024;
static constexpr int H_  = 16;
static constexpr int HD_ = 64;
static constexpr int BT_ = B_ * T_;       // 8192 rows
static constexpr float LN_THETA = 9.210340371976184f; // ln(10000)

// ---------- CDNA5 async global->LDS staging ----------
#if defined(__has_builtin)
#if __has_builtin(__builtin_amdgcn_s_wait_asynccnt)
#define WAIT_ASYNC(n) __builtin_amdgcn_s_wait_asynccnt(n)
#endif
#endif
#ifndef WAIT_ASYNC
#define WAIT_ASYNC(n) asm volatile("s_wait_asynccnt %0" ::"n"(n) : "memory")
#endif

// One b128 (16B) async transfer per lane: LDS[lds] = MEM[g..g+15]; tracked by ASYNCcnt.
// Low 32 bits of a flat pointer into LDS == the LDS byte offset (AS3->flat keeps it).
__device__ __forceinline__ void async_b128(const uint16_t* g, uint16_t* l) {
  uint32_t lds = (uint32_t)(uintptr_t)l;
  uint64_t ga  = (uint64_t)(uintptr_t)g;
  asm volatile("global_load_async_to_lds_b128 %0, %1, off"
               :: "v"(lds), "v"(ga)
               : "memory");
}

__device__ __forceinline__ uint16_t f2bf(float f) {
  union { float f; uint32_t u; } c; c.f = f;
  uint32_t r = c.u + 0x7FFFu + ((c.u >> 16) & 1u);   // round-to-nearest-even
  return (uint16_t)(r >> 16);
}

union FragU { u32x4 q[2]; v16bf v; };

// A fragment: 16x32 bf16 (MxK), row-major source with row stride `stride` elems.
__device__ __forceinline__ v16bf load_a_frag(const uint16_t* p, int stride) {
  int lane = threadIdx.x & 31;
  int row  = lane & 15;
  int k0   = (lane >> 4) * 8;
  FragU f;
  f.q[0] = *(const u32x4*)(p + row * stride + k0);
  f.q[1] = *(const u32x4*)(p + row * stride + k0 + 16);
  return f.v;
}

// B fragment: 32x16 bf16 (KxN), source stored [N][K] row-major (stride elems).
__device__ __forceinline__ v16bf load_b_frag(const uint16_t* p, int stride) {
  int lane = threadIdx.x & 31;
  int col  = lane & 15;
  int kh   = (lane >> 4) * 8;
  FragU f;
  f.q[0] = *(const u32x4*)(p + col * stride + kh);
  f.q[1] = *(const u32x4*)(p + col * stride + kh + 16);
  return f.v;
}

// ---------- kernel 1: f32 -> bf16 cast ----------
__global__ void cvt_f32_bf16_kernel(const float* __restrict__ src,
                                    uint16_t* __restrict__ dst, int n) {
  int i = (blockIdx.x * blockDim.x + threadIdx.x) * 4;
  if (i + 3 < n) {
    float a0 = src[i], a1 = src[i + 1], a2 = src[i + 2], a3 = src[i + 3];
    uint32_t lo = (uint32_t)f2bf(a0) | ((uint32_t)f2bf(a1) << 16);
    uint32_t hi = (uint32_t)f2bf(a2) | ((uint32_t)f2bf(a3) << 16);
    *(u32a*)(dst + i)     = lo;
    *(u32a*)(dst + i + 2) = hi;
  } else {
    for (; i < n; ++i) dst[i] = f2bf(src[i]);
  }
}

// ---------- shared GEMM core (double-buffered async staging) ----------
// Block 256 thr / 8 waves; block tile M=128, N=64; wave = 32x32 (2x2 WMMA tiles).
// Per tile, per thread: 3 async b128 ops (2 A rows, 1 B row).
__device__ __forceinline__ void stage_gemm_tile(const uint16_t* __restrict__ abase,
                                                const uint16_t* __restrict__ bbase,
                                                int m0, int n0, int kk,
                                                uint16_t* AshBuf, uint16_t* BshBuf) {
  int u   = threadIdx.x;
  int row = u >> 2, c = (u & 3) * 8;   // row 0..63, c in {0,8,16,24}
  async_b128(&abase[(size_t)(m0 + row) * D_ + kk + c],      &AshBuf[row * 32 + c]);
  async_b128(&abase[(size_t)(m0 + row + 64) * D_ + kk + c], &AshBuf[(row + 64) * 32 + c]);
  async_b128(&bbase[(size_t)(n0 + row) * D_ + kk + c],      &BshBuf[row * 32 + c]);
}

template <typename Epilogue>
__device__ __forceinline__ void gemm_128x64(const uint16_t* __restrict__ abase,
                                            const uint16_t* __restrict__ bbase,
                                            int m0, int n0, Epilogue epi) {
  __shared__ uint16_t Ash[2][128 * 32];
  __shared__ uint16_t Bsh[2][64 * 32];

  const int w  = threadIdx.x >> 5;
  const int wm = w & 3, wn = w >> 2;

  v8f z = {0.f, 0.f, 0.f, 0.f, 0.f, 0.f, 0.f, 0.f};
  v8f acc[2][2] = {{z, z}, {z, z}};

  stage_gemm_tile(abase, bbase, m0, n0, 0, Ash[0], Bsh[0]);
  for (int kk = 0; kk < D_; kk += 32) {
    const int  cur  = (kk >> 5) & 1;
    const bool more = (kk + 32) < D_;
    if (more) {
      stage_gemm_tile(abase, bbase, m0, n0, kk + 32, Ash[1 - cur], Bsh[1 - cur]);
      WAIT_ASYNC(3);               // previous tile's 3 ops complete (in-order)
    } else {
      WAIT_ASYNC(0);
    }
    __syncthreads();
    v16bf a0 = load_a_frag(&Ash[cur][(wm * 32) * 32], 32);
    v16bf a1 = load_a_frag(&Ash[cur][(wm * 32 + 16) * 32], 32);
    v16bf b0 = load_b_frag(&Bsh[cur][(wn * 32) * 32], 32);
    v16bf b1 = load_b_frag(&Bsh[cur][(wn * 32 + 16) * 32], 32);
    acc[0][0] = WMMA_BF16(a0, b0, acc[0][0]);
    acc[0][1] = WMMA_BF16(a0, b1, acc[0][1]);
    acc[1][0] = WMMA_BF16(a1, b0, acc[1][0]);
    acc[1][1] = WMMA_BF16(a1, b1, acc[1][1]);
    __syncthreads();               // keep next stage's LDS writes off live buffer
  }
  epi(acc, wm, wn);
}

// ---------- kernel 2: qkv = x @ Wqkv^T, fused RoPE ----------
// q,k bf16 [B,H,T,hd]; v bf16 TRANSPOSED [B,H,hd,T] for the attention PV gemm.
__global__ __launch_bounds__(256) void qkv_rope_kernel(
    const uint16_t* __restrict__ xbf,   // [8192][1024]
    const uint16_t* __restrict__ wqkv,  // [3072][1024]
    uint16_t* __restrict__ qbf, uint16_t* __restrict__ kbf,
    uint16_t* __restrict__ vtbf) {
  const int m0 = blockIdx.x * 128;
  const int n0 = blockIdx.y * 64;
  const int lane = threadIdx.x & 31;
  const int colL = lane & 15;
  const int rowOff = (lane >> 4) * 8;

  gemm_128x64(xbf, wqkv, m0, n0, [&](v8f (&acc)[2][2], int wm, int wn) {
#pragma unroll
    for (int i = 0; i < 2; ++i)
#pragma unroll
      for (int j = 0; j < 2; ++j) {
        int ng  = n0 + wn * 32 + j * 16 + colL;
        int sec = ng >> 10;          // 0=q 1=k 2=v (uniform per 16-col tile)
        int nn  = ng & 1023;
        int h   = nn >> 6;
        int d   = nn & 63;
#pragma unroll
        for (int r = 0; r < 8; ++r) {
          int rg = m0 + wm * 32 + i * 16 + r + rowOff;
          int bi = rg >> 10;         // batch
          int t  = rg & 1023;        // position
          float v = acc[i][j][r];
          if (sec < 2) {             // RoPE for q,k
            float vo = __shfl_xor(v, 1);
            int   fi = d >> 1;
            float ang = (float)t * __expf(-((float)(2 * fi)) * (LN_THETA / 64.0f));
            float cs = __cosf(ang), sn = __sinf(ang);
            v = (d & 1) ? (v * cs + vo * sn) : (v * cs - vo * sn);
            if (sec == 0) v *= 0.125f;   // 1/sqrt(64)
          }
          uint16_t bv = f2bf(v);
          if (sec == 0)
            qbf[(((size_t)bi * H_ + h) * T_ + t) * HD_ + d] = bv;
          else if (sec == 1)
            kbf[(((size_t)bi * H_ + h) * T_ + t) * HD_ + d] = bv;
          else  // transposed V: [B,H,hd,T]
            vtbf[(((size_t)bi * H_ + h) * HD_ + d) * T_ + t] = bv;
        }
      }
  });
}

// ---------- kernel 3: flash attention (double-buffered async K/V tiles) ----------
static constexpr int VT_STRIDE = 40;  // padded Vt row, 80B (16B multiple)
__global__ __launch_bounds__(256) void attn_kernel(
    const uint16_t* __restrict__ qbf, const uint16_t* __restrict__ kbf,
    const uint16_t* __restrict__ vtbf, const uint8_t* __restrict__ mask,
    uint16_t* __restrict__ pre_out) {
  __shared__ uint16_t Ksh[2][32 * 64];            // K tile [key][d]
  __shared__ uint16_t Vt[2][64 * VT_STRIDE];      // V^T tile [d][key]
  __shared__ uint16_t Psh[8][16 * 32];            // per-wave P staging

  const int bh = blockIdx.x;
  const int b  = bh >> 4, h = bh & 15;
  const int w  = threadIdx.x >> 5;
  const int lane = threadIdx.x & 31;
  const int colL = lane & 15;
  const int q0 = blockIdx.y * 128 + w * 16;

  const uint16_t* qrow   = qbf + ((size_t)bh * T_ + q0) * HD_;
  const uint16_t* kbase  = kbf + (size_t)bh * T_ * HD_;
  const uint16_t* vtbase = vtbf + (size_t)bh * HD_ * T_;
  const uint8_t*  mrow_g = mask + (size_t)b * T_;

  const v16bf aq0 = load_a_frag(qrow, HD_);       // d 0..31
  const v16bf aq1 = load_a_frag(qrow + 32, HD_);  // d 32..63

  v8f z = {0.f, 0.f, 0.f, 0.f, 0.f, 0.f, 0.f, 0.f};
  v8f o[4] = {z, z, z, z};
  float mrow[8], lrow[8];
#pragma unroll
  for (int r = 0; r < 8; ++r) { mrow[r] = -3.0e38f; lrow[r] = 0.f; }

  // per-tile staging: 2 async b128 ops per thread (1 K, 1 Vt)
  auto stage = [&](int kt, int buf) {
    int u = threadIdx.x;
    int kr = u >> 3, kc = (u & 7) * 8;            // K: 32 x 64
    async_b128(&kbase[(size_t)(kt + kr) * HD_ + kc], &Ksh[buf][kr * 64 + kc]);
    int vr = u >> 2, vc = (u & 3) * 8;            // Vt: 64 x 32
    async_b128(&vtbase[(size_t)vr * T_ + kt + vc], &Vt[buf][vr * VT_STRIDE + vc]);
  };

  stage(0, 0);
  for (int kt = 0; kt < T_; kt += 32) {
    const int  cur  = (kt >> 5) & 1;
    const bool more = (kt + 32) < T_;
    if (more) {
      stage(kt + 32, 1 - cur);
      WAIT_ASYNC(2);
    } else {
      WAIT_ASYNC(0);
    }
    __syncthreads();

    // scores (1/sqrt(hd) folded into q)
    v8f s0 = WMMA_BF16(aq0, load_b_frag(&Ksh[cur][0], 64), z);
    s0     = WMMA_BF16(aq1, load_b_frag(&Ksh[cur][32], 64), s0);
    v8f s1 = WMMA_BF16(aq0, load_b_frag(&Ksh[cur][16 * 64], 64), z);
    s1     = WMMA_BF16(aq1, load_b_frag(&Ksh[cur][16 * 64 + 32], 64), s1);

    const float ma0 = mrow_g[kt + colL]      ? 0.0f : -3.0e38f;
    const float ma1 = mrow_g[kt + 16 + colL] ? 0.0f : -3.0e38f;
#pragma unroll
    for (int r = 0; r < 8; ++r) {
      float x0 = s0[r] + ma0;
      float x1 = s1[r] + ma1;
      float mx = fmaxf(x0, x1);
#pragma unroll
      for (int dd = 1; dd < 16; dd <<= 1) mx = fmaxf(mx, __shfl_xor(mx, dd));
      float mnew  = fmaxf(mrow[r], mx);
      float alpha = __expf(mrow[r] - mnew);
      float p0 = __expf(x0 - mnew), p1 = __expf(x1 - mnew);
      float sum = p0 + p1;
#pragma unroll
      for (int dd = 1; dd < 16; dd <<= 1) sum += __shfl_xor(sum, dd);
      lrow[r] = lrow[r] * alpha + sum;
      mrow[r] = mnew;
#pragma unroll
      for (int j = 0; j < 4; ++j) o[j][r] *= alpha;
      int prow = r + ((lane >> 4) * 8);
      Psh[w][prow * 32 + colL]      = f2bf(p0);
      Psh[w][prow * 32 + 16 + colL] = f2bf(p1);
    }

    // O += P @ V  (same-wave LDS ops are in-order; no barrier needed)
    v16bf pa = load_a_frag(Psh[w], 32);
#pragma unroll
    for (int j = 0; j < 4; ++j)
      o[j] = WMMA_BF16(pa, load_b_frag(&Vt[cur][(j * 16) * VT_STRIDE], VT_STRIDE), o[j]);

    __syncthreads();
  }

  // epilogue: O /= l, write bf16 pre_out[B,T,D]
  const int rowOff = (lane >> 4) * 8;
#pragma unroll
  for (int r = 0; r < 8; ++r) {
    float inv = 1.0f / lrow[r];
    int qg = q0 + r + rowOff;
    size_t base = ((size_t)b * T_ + qg) * D_ + (size_t)h * HD_;
#pragma unroll
    for (int j = 0; j < 4; ++j)
      pre_out[base + j * 16 + colL] = f2bf(o[j][r] * inv);
  }
}

// ---------- kernel 4: out = pre_out @ Wout^T + bout (f32 out) ----------
__global__ __launch_bounds__(256) void out_proj_kernel(
    const uint16_t* __restrict__ abf,   // [8192][1024] bf16
    const uint16_t* __restrict__ wout,  // [1024][1024] bf16
    const float* __restrict__ bout, float* __restrict__ out) {
  const int m0 = blockIdx.x * 128;
  const int n0 = blockIdx.y * 64;
  const int lane = threadIdx.x & 31;
  const int colL = lane & 15;
  const int rowOff = (lane >> 4) * 8;

  gemm_128x64(abf, wout, m0, n0, [&](v8f (&acc)[2][2], int wm, int wn) {
#pragma unroll
    for (int i = 0; i < 2; ++i)
#pragma unroll
      for (int j = 0; j < 2; ++j) {
        int ng = n0 + wn * 32 + j * 16 + colL;
        float bias = bout[ng];
#pragma unroll
        for (int r = 0; r < 8; ++r) {
          int rg = m0 + wm * 32 + i * 16 + r + rowOff;
          out[(size_t)rg * D_ + ng] = acc[i][j][r] + bias;
        }
      }
  });
}

// ---------- host ----------
extern "C" void kernel_launch(void* const* d_in, const int* in_sizes, int n_in,
                              void* d_out, int out_size, void* d_ws, size_t ws_size,
                              hipStream_t stream) {
  const float*   x    = (const float*)d_in[0];
  const uint8_t* mask = (const uint8_t*)d_in[1];
  const float*   Wqkv = (const float*)d_in[2];
  const float*   Wout = (const float*)d_in[3];
  const float*   bout = (const float*)d_in[4];
  float* out = (float*)d_out;

  char* ws = (char*)d_ws;
  size_t ofs = 0;
  auto carve = [&](size_t bytes) {
    char* p = ws + ofs;
    ofs += (bytes + 255) & ~(size_t)255;
    return p;
  };
  uint16_t* xbf    = (uint16_t*)carve((size_t)BT_ * D_ * 2);       // 16 MB
  uint16_t* wqkvbf = (uint16_t*)carve((size_t)3 * D_ * D_ * 2);    //  6 MB
  uint16_t* woutbf = (uint16_t*)carve((size_t)D_ * D_ * 2);        //  2 MB
  uint16_t* qbf    = (uint16_t*)carve((size_t)BT_ * D_ * 2);       // 16 MB
  uint16_t* kbf    = (uint16_t*)carve((size_t)BT_ * D_ * 2);       // 16 MB
  uint16_t* vtbf   = (uint16_t*)carve((size_t)BT_ * D_ * 2);       // 16 MB
  uint16_t* pre    = xbf;  // alias: x is dead after the QKV GEMM

  const int nX = BT_ * D_, nWq = 3 * D_ * D_, nWo = D_ * D_;
  cvt_f32_bf16_kernel<<<(nX / 4 + 255) / 256, 256, 0, stream>>>(x, xbf, nX);
  cvt_f32_bf16_kernel<<<(nWq / 4 + 255) / 256, 256, 0, stream>>>(Wqkv, wqkvbf, nWq);
  cvt_f32_bf16_kernel<<<(nWo / 4 + 255) / 256, 256, 0, stream>>>(Wout, woutbf, nWo);

  qkv_rope_kernel<<<dim3(BT_ / 128, (3 * D_) / 64), 256, 0, stream>>>(
      xbf, wqkvbf, qbf, kbf, vtbf);

  attn_kernel<<<dim3(B_ * H_, T_ / 128), 256, 0, stream>>>(
      qbf, kbf, vtbf, mask, pre);

  out_proj_kernel<<<dim3(BT_ / 128, D_ / 64), 256, 0, stream>>>(
      pre, woutbf, bout, out);
}
